// LlamaAttention_hqe_10763188044285
// MI455X (gfx1250) — matrix operand order repaired and checked by reading, not compiled, not run on hardware
//
#include <hip/hip_runtime.h>

#define S_LEN 2048
#define HIDN  4096
#define NH    32
#define NKVH  8
#define HD    128
#define LASTQ 20

typedef __attribute__((ext_vector_type(16))) __bf16        v16bf;
typedef __attribute__((ext_vector_type(8)))  float         v8f;
typedef __attribute__((ext_vector_type(4)))  unsigned int  uint4v;

union FragAB { uint4v q[2]; v16bf v; };

__device__ __forceinline__ unsigned short f32_to_bf16(float f) {
  unsigned int u = __float_as_uint(f);
  u += 0x7FFFu + ((u >> 16) & 1u);
  return (unsigned short)(u >> 16);
}
__device__ __forceinline__ float bf16_to_f32(unsigned short h) {
  return __uint_as_float(((unsigned int)h) << 16);
}

// CDNA5 async copy: global -> LDS, tracked on ASYNCcnt (ISA 08_async_tensor §4)
__device__ __forceinline__ void async_load_b128(unsigned lds_addr, const void* gptr) {
  asm volatile("global_load_async_to_lds_b128 %0, %1, off"
               :: "v"(lds_addr), "v"((unsigned long long)(uintptr_t)gptr)
               : "memory");
}
__device__ __forceinline__ void wait_async0() {
  asm volatile("s_wait_asynccnt 0x0" ::: "memory");
}
__device__ __forceinline__ unsigned lds_off(const void* p) {
  return (unsigned)(uintptr_t)p;   // shared-aperture flat addr: low 32 bits = LDS byte addr
}

// ---------------------------------------------------------------- utilities
__global__ void f2bf_kernel(const float* __restrict__ in,
                            unsigned short* __restrict__ out, int n) {
  int i = blockIdx.x * 256 + threadIdx.x;
  if (i < n) out[i] = f32_to_bf16(in[i]);
}

__global__ void zero_kernel(float* p, int n) {
  int i = blockIdx.x * 256 + threadIdx.x;
  if (i < n) p[i] = 0.f;
}

// ---------------------------------------------------------------- WMMA GEMM
// C[M][N] (f32) = A[M][K] (bf16 rowmajor) * B[K][N] (bf16 rowmajor)
// workgroup tile 128x128, 8 waves -> 4x2 grid of 32x64 wave tiles, BK=32
__global__ __launch_bounds__(256) void gemm_bf16_kernel(
    const unsigned short* __restrict__ A, const unsigned short* __restrict__ B,
    float* __restrict__ C, int M, int N, int K) {
  __shared__ unsigned short Alds[128][32];  // row-major tile (async-filled)
  __shared__ unsigned short Blds[128][32];  // transposed tile: [n][k]
  int tid = threadIdx.x;
  int wave = tid >> 5, lane = tid & 31;
  int lo = lane & 15, hi = lane >> 4;
  int wm = (wave >> 1) * 32;
  int wn = (wave & 1) * 64;
  int mbase = blockIdx.x * 128;
  int nbase = blockIdx.y * 128;

  v8f zero8 = {0.f,0.f,0.f,0.f,0.f,0.f,0.f,0.f};
  v8f acc[2][4];
#pragma unroll
  for (int mt = 0; mt < 2; ++mt)
#pragma unroll
    for (int nt = 0; nt < 4; ++nt) acc[mt][nt] = zero8;

  int ar = tid >> 1, ac = (tid & 1) * 16;       // A tile loader: 128x32
  int bkr = tid >> 3, bc = (tid & 7) * 16;      // B tile loader: 32x128

  for (int k0 = 0; k0 < K; k0 += 32) {
    __syncthreads();
    {  // A tile: async global->LDS, no VGPR round trip
      const unsigned short* ga = A + (size_t)(mbase + ar) * K + k0 + ac;
      unsigned la = lds_off(&Alds[ar][ac]);
      async_load_b128(la, ga);
      async_load_b128(la + 16, ga + 8);
    }
    {  // B tile: sync load + transposed store into LDS
      union { uint4v q[2]; unsigned short s[16]; } u;
      const uint4v* src = (const uint4v*)(B + (size_t)(k0 + bkr) * N + nbase + bc);
      u.q[0] = src[0]; u.q[1] = src[1];
      if (k0 + 32 < K)
        __builtin_prefetch(B + (size_t)(k0 + 32 + bkr) * N + nbase + bc, 0, 1);
#pragma unroll
      for (int e = 0; e < 16; ++e) Blds[bc + e][bkr] = u.s[e];
    }
    wait_async0();
    __syncthreads();

    FragAB afr[2], bfr[4];
#pragma unroll
    for (int mt = 0; mt < 2; ++mt) {
      const unsigned short* p = &Alds[wm + 16 * mt + lo][0];
      afr[mt].q[0] = *(const uint4v*)(p + hi * 8);
      afr[mt].q[1] = *(const uint4v*)(p + hi * 8 + 16);
    }
#pragma unroll
    for (int nt = 0; nt < 4; ++nt) {
      const unsigned short* p = &Blds[wn + 16 * nt + lo][0];
      bfr[nt].q[0] = *(const uint4v*)(p + hi * 8);
      bfr[nt].q[1] = *(const uint4v*)(p + hi * 8 + 16);
    }
#pragma unroll
    for (int mt = 0; mt < 2; ++mt)
#pragma unroll
      for (int nt = 0; nt < 4; ++nt)
        acc[mt][nt] = __builtin_amdgcn_wmma_f32_16x16x32_bf16(
            false, afr[mt].v, false, bfr[nt].v, (short)0, acc[mt][nt], false, false);
  }

#pragma unroll
  for (int mt = 0; mt < 2; ++mt)
#pragma unroll
    for (int nt = 0; nt < 4; ++nt)
#pragma unroll
      for (int r = 0; r < 8; ++r) {
        int row = mbase + wm + 16 * mt + r + hi * 8;
        int col = nbase + wn + 16 * nt + lo;
        C[(size_t)row * N + col] = acc[mt][nt][r];
      }
}

// ---------------------------------------------------------------- RoPE
__global__ void rope_q_kernel(const float* __restrict__ Q,
                              const float* __restrict__ cosb,
                              const float* __restrict__ sinb,
                              unsigned short* __restrict__ Qb) {
  int idx = blockIdx.x * 256 + threadIdx.x;           // S*NH*HD threads
  if (idx >= S_LEN * NH * HD) return;
  int d = idx & 127, h = (idx >> 7) & 31, s = idx >> 12;
  size_t base = (size_t)s * (NH * HD) + h * HD;
  float x = Q[base + d];
  float o = (d < 64) ? -Q[base + d + 64] : Q[base + d - 64];
  float r = x * cosb[s * HD + d] + o * sinb[s * HD + d];
  Qb[((size_t)h * S_LEN + s) * HD + d] = f32_to_bf16(r);
}

__global__ void rope_k_kernel(const float* __restrict__ Kin,
                              const float* __restrict__ cosb,
                              const float* __restrict__ sinb,
                              float* __restrict__ Kr) {
  int idx = blockIdx.x * 256 + threadIdx.x;           // S*NKVH*HD threads
  if (idx >= S_LEN * NKVH * HD) return;
  int d = idx & 127, kv = (idx >> 7) & 7, s = idx >> 10;
  size_t base = (size_t)s * (NKVH * HD) + kv * HD;
  float x = Kin[base + d];
  float o = (d < 64) ? -Kin[base + d + 64] : Kin[base + d - 64];
  float r = x * cosb[s * HD + d] + o * sinb[s * HD + d];
  Kr[((size_t)kv * S_LEN + s) * HD + d] = r;
}

// ------------------------------------------------------- importance scoring
__global__ __launch_bounds__(256) void importance_kernel(
    const unsigned short* __restrict__ Qb, const float* __restrict__ Kr,
    float* __restrict__ TI) {
  const float scaling = 0.08838834764831845f;
  int h = blockIdx.y, qi = blockIdx.x;
  int sq = S_LEN - LASTQ + qi;
  int kv = h >> 2;
  int tid = threadIdx.x;
  __shared__ float qs[HD];
  __shared__ float red[256];
  if (tid < HD) qs[tid] = bf16_to_f32(Qb[((size_t)h * S_LEN + sq) * HD + tid]);
  __syncthreads();

  float sc[8], lmax = -1e30f;
#pragma unroll
  for (int j0 = 0; j0 < 8; ++j0) {
    int j = tid + j0 * 256;
    const float* krow = Kr + ((size_t)kv * S_LEN + j) * HD;
    float dot = 0.f;
    for (int d = 0; d < HD; ++d) dot += qs[d] * krow[d];
    float v = dot * scaling;
    if (j > sq) v = -1e30f;
    sc[j0] = v; lmax = fmaxf(lmax, v);
  }
  red[tid] = lmax; __syncthreads();
  for (int s2 = 128; s2 > 0; s2 >>= 1) {
    if (tid < s2) red[tid] = fmaxf(red[tid], red[tid + s2]);
    __syncthreads();
  }
  float bmax = red[0]; __syncthreads();
  float lsum = 0.f;
#pragma unroll
  for (int j0 = 0; j0 < 8; ++j0) { sc[j0] = __expf(sc[j0] - bmax); lsum += sc[j0]; }
  red[tid] = lsum; __syncthreads();
  for (int s2 = 128; s2 > 0; s2 >>= 1) {
    if (tid < s2) red[tid] += red[tid + s2];
    __syncthreads();
  }
  float inv = 1.f / red[0];
#pragma unroll
  for (int j0 = 0; j0 < 8; ++j0) {
    int j = tid + j0 * 256;
    float cnt = (j < S_LEN - LASTQ) ? (float)LASTQ : (float)(S_LEN - j);
    atomicAdd(&TI[j], sc[j0] * inv / cnt);
  }
}

// --------------------------------------------------- block sort + bit alloc
__global__ void bits_kernel(const float* __restrict__ TI,
                            const float* __restrict__ ratio,
                            float* __restrict__ levels) {
  __shared__ float bimp[64];
  __shared__ int order[64];
  __shared__ int kb[4];
  int tid = threadIdx.x;
  if (tid < 64) {
    float s = 0.f;
    for (int i = 0; i < 32; ++i) s += TI[tid * 32 + i];
    bimp[tid] = s;
  }
  __syncthreads();
  if (tid < 64) {
    float mv = bimp[tid];
    int rank = 0;
    for (int o = 0; o < 64; ++o) {
      float ov = bimp[o];
      if (ov > mv || (ov == mv && o < tid)) rank++;
    }
    order[rank] = tid;
  }
  if (tid == 0) {
    int k0 = (int)rintf(64.f * ratio[0]);
    int k1 = (int)rintf(64.f * ratio[1]);
    int k2 = (int)rintf(64.f * ratio[2]);
    kb[0] = k0; kb[1] = k0 + k1; kb[2] = k0 + k1 + k2; kb[3] = 64;
  }
  __syncthreads();
  if (tid < 64) {
    int p = tid, g = 0;
    while (g < 3 && p >= kb[g]) g++;
    int bits = (g == 0) ? 8 : (g == 1) ? 4 : (g == 2) ? 2 : 1;
    float lev = (float)((1 << bits) - 1);
    int blk = order[p];
    for (int i = 0; i < 32; ++i) levels[blk * 32 + i] = lev;
  }
}

// ------------------------------------------------------------- fake quant
__global__ __launch_bounds__(128) void quant_kernel(
    const float* __restrict__ Kr, const float* __restrict__ Vf,
    const float* __restrict__ levels,
    unsigned short* __restrict__ Kq, unsigned short* __restrict__ Vq) {
  int bid = blockIdx.x;                // NKVH*S blocks
  int kv = bid >> 11, s = bid & 2047;
  int d = threadIdx.x;
  float kval = Kr[((size_t)kv * S_LEN + s) * HD + d];
  float vval = Vf[(size_t)s * (NKVH * HD) + kv * HD + d];
  float kmn = kval, kmx = kval, vmn = vval, vmx = vval;
  for (int m = 16; m >= 1; m >>= 1) {
    kmn = fminf(kmn, __shfl_xor(kmn, m, 32));
    kmx = fmaxf(kmx, __shfl_xor(kmx, m, 32));
    vmn = fminf(vmn, __shfl_xor(vmn, m, 32));
    vmx = fmaxf(vmx, __shfl_xor(vmx, m, 32));
  }
  __shared__ float r4[4][4];
  int w = d >> 5;
  if ((d & 31) == 0) { r4[0][w] = kmn; r4[1][w] = kmx; r4[2][w] = vmn; r4[3][w] = vmx; }
  __syncthreads();
  kmn = fminf(fminf(r4[0][0], r4[0][1]), fminf(r4[0][2], r4[0][3]));
  kmx = fmaxf(fmaxf(r4[1][0], r4[1][1]), fmaxf(r4[1][2], r4[1][3]));
  vmn = fminf(fminf(r4[2][0], r4[2][1]), fminf(r4[2][2], r4[2][3]));
  vmx = fmaxf(fmaxf(r4[3][0], r4[3][1]), fmaxf(r4[3][2], r4[3][3]));
  float lev = levels[s];
  float ks = fmaxf((kmx - kmn) / lev, 1e-8f);
  float vs = fmaxf((vmx - vmn) / lev, 1e-8f);
  float kq = rintf((kval - kmn) / ks) * ks + kmn;
  float vq = rintf((vval - vmn) / vs) * vs + vmn;
  Kq[((size_t)kv * S_LEN + s) * HD + d] = f32_to_bf16(kq);
  Vq[((size_t)kv * S_LEN + s) * HD + d] = f32_to_bf16(vq);
}

// ------------------------------------------------- flash attention (causal)
// grid (S/128, NH), 256 threads = 8 waves, each wave owns 16 query rows.
__global__ __launch_bounds__(256) void attn_kernel(
    const unsigned short* __restrict__ Qb, const unsigned short* __restrict__ Kq,
    const unsigned short* __restrict__ Vq, unsigned short* __restrict__ Ob) {
  const float scaling = 0.08838834764831845f;
  __shared__ unsigned short Klds[32 * 128];   // [key][dim] (async-filled)
  __shared__ unsigned short Vt[128 * 32];     // [dim][key]
  __shared__ unsigned short Plds[8 * 16 * 32];
  int tid = threadIdx.x, wave = tid >> 5, lane = tid & 31;
  int lo = lane & 15, hi = lane >> 4;
  int h = blockIdx.y, kv = h >> 2;
  int q0 = blockIdx.x * 128 + wave * 16;

  FragAB aq[4];
  {
    const unsigned short* qp = Qb + ((size_t)h * S_LEN + q0 + lo) * HD + hi * 8;
#pragma unroll
    for (int ks = 0; ks < 4; ++ks) {
      aq[ks].q[0] = *(const uint4v*)(qp + 32 * ks);
      aq[ks].q[1] = *(const uint4v*)(qp + 32 * ks + 16);
    }
  }
  v8f zero8 = {0.f,0.f,0.f,0.f,0.f,0.f,0.f,0.f};
  v8f o[8];
  float m[8], l[8];
#pragma unroll
  for (int r = 0; r < 8; ++r) { o[r] = zero8; m[r] = -1e30f; l[r] = 0.f; }

  int ldkey = tid >> 3, ldc = (tid & 7) * 16;
  int nkb = blockIdx.x * 4 + 4;
  for (int kb = 0; kb < nkb; ++kb) {
    int kbase = kb * 32;
    __syncthreads();
    {  // K tile: async global->LDS
      const unsigned short* gk = Kq + ((size_t)kv * S_LEN + kbase + ldkey) * HD + ldc;
      unsigned lk = lds_off(&Klds[ldkey * 128 + ldc]);
      async_load_b128(lk, gk);
      async_load_b128(lk + 16, gk + 8);
    }
    {  // V tile: sync load + transposed store
      union { uint4v q[2]; unsigned short s[16]; } u;
      const uint4v* src = (const uint4v*)(Vq + ((size_t)kv * S_LEN + kbase + ldkey) * HD + ldc);
      u.q[0] = src[0]; u.q[1] = src[1];
#pragma unroll
      for (int e = 0; e < 16; ++e) Vt[(ldc + e) * 32 + ldkey] = u.s[e];
    }
    wait_async0();
    __syncthreads();
    if (kbase <= q0 + 15) {                  // wave-uniform: EXEC stays all-1s
      v8f sacc[2] = {zero8, zero8};
#pragma unroll
      for (int nt = 0; nt < 2; ++nt) {
        const unsigned short* kp = &Klds[(16 * nt + lo) * 128];
#pragma unroll
        for (int ks = 0; ks < 4; ++ks) {
          FragAB bk;
          int cb = hi * 8 + 32 * ks;
          bk.q[0] = *(const uint4v*)(kp + cb);
          bk.q[1] = *(const uint4v*)(kp + cb + 16);
          sacc[nt] = __builtin_amdgcn_wmma_f32_16x16x32_bf16(
              false, aq[ks].v, false, bk.v, (short)0, sacc[nt], false, false);
        }
      }
      unsigned short* pw = &Plds[wave * 512];
#pragma unroll
      for (int r = 0; r < 8; ++r) {
        int row = q0 + r + hi * 8;
        float v0 = sacc[0][r] * scaling;
        float v1 = sacc[1][r] * scaling;
        if (kbase + lo > row)      v0 = -1e30f;
        if (kbase + 16 + lo > row) v1 = -1e30f;
        float rmax = fmaxf(v0, v1);
        for (int mk = 1; mk <= 8; mk <<= 1) rmax = fmaxf(rmax, __shfl_xor(rmax, mk, 32));
        float mnew = fmaxf(m[r], rmax);
        float alpha = __expf(m[r] - mnew);
        float p0 = __expf(v0 - mnew), p1 = __expf(v1 - mnew);
        float rsum = p0 + p1;
        for (int mk = 1; mk <= 8; mk <<= 1) rsum += __shfl_xor(rsum, mk, 32);
        l[r] = l[r] * alpha + rsum;
        m[r] = mnew;
#pragma unroll
        for (int dt = 0; dt < 8; ++dt) o[dt][r] *= alpha;
        int prow = r + hi * 8;
        pw[prow * 32 + lo] = f32_to_bf16(p0);
        pw[prow * 32 + 16 + lo] = f32_to_bf16(p1);
      }
      FragAB pf;
      pf.q[0] = *(const uint4v*)(&pw[lo * 32 + hi * 8]);
      pf.q[1] = *(const uint4v*)(&pw[lo * 32 + hi * 8 + 16]);
#pragma unroll
      for (int dt = 0; dt < 8; ++dt) {
        FragAB bv;
        const unsigned short* vp = &Vt[(16 * dt + lo) * 32 + hi * 8];
        bv.q[0] = *(const uint4v*)(vp);
        bv.q[1] = *(const uint4v*)(vp + 16);
        o[dt] = __builtin_amdgcn_wmma_f32_16x16x32_bf16(
            false, pf.v, false, bv.v, (short)0, o[dt], false, false);
      }
    }
  }
#pragma unroll
  for (int r = 0; r < 8; ++r) {
    int row = q0 + r + hi * 8;
    float inv = 1.0f / l[r];
#pragma unroll
    for (int dt = 0; dt < 8; ++dt)
      Ob[(size_t)row * (NH * HD) + h * HD + 16 * dt + lo] = f32_to_bf16(o[dt][r] * inv);
  }
}

// ---------------------------------------------------------------- launcher
extern "C" void kernel_launch(void* const* d_in, const int* in_sizes, int n_in,
                              void* d_out, int out_size, void* d_ws, size_t ws_size,
                              hipStream_t stream) {
  const float* X    = (const float*)d_in[0];
  const float* cosb = (const float*)d_in[1];
  const float* sinb = (const float*)d_in[2];
  // d_in[3] = attention_mask (causal; applied analytically)
  const float* ratio = (const float*)d_in[4];
  const float* Wq = (const float*)d_in[5];
  const float* Wk = (const float*)d_in[6];
  const float* Wv = (const float*)d_in[7];
  const float* Wo = (const float*)d_in[8];
  float* out = (float*)d_out;

  char* w = (char*)d_ws;
  size_t off = 0;
  auto nxt = [&](size_t b) { void* p = w + off; off += (b + 255) & ~(size_t)255; return p; };
  unsigned short* Xb  = (unsigned short*)nxt((size_t)S_LEN * HIDN * 2);
  unsigned short* Wb  = (unsigned short*)nxt((size_t)HIDN * HIDN * 2);
  float*          Qf  = (float*)nxt((size_t)S_LEN * HIDN * 4);
  float*          Kf  = (float*)nxt((size_t)S_LEN * NKVH * HD * 4);
  float*          Vf  = (float*)nxt((size_t)S_LEN * NKVH * HD * 4);
  unsigned short* Qbb = (unsigned short*)nxt((size_t)S_LEN * HIDN * 2);
  float*          Krf = (float*)nxt((size_t)S_LEN * NKVH * HD * 4);
  unsigned short* Kqb = (unsigned short*)nxt((size_t)S_LEN * NKVH * HD * 2);
  unsigned short* Vqb = (unsigned short*)nxt((size_t)S_LEN * NKVH * HD * 2);
  unsigned short* Obb = (unsigned short*)nxt((size_t)S_LEN * HIDN * 2);
  float*          TI  = (float*)nxt(S_LEN * 4);
  float*          LEV = (float*)nxt(S_LEN * 4);

  const int nX  = S_LEN * HIDN;
  const int nWq = HIDN * HIDN;
  const int nWk = HIDN * NKVH * HD;

  f2bf_kernel<<<(nX + 255) / 256, 256, 0, stream>>>(X, Xb, nX);

  // Q = X @ Wq
  f2bf_kernel<<<(nWq + 255) / 256, 256, 0, stream>>>(Wq, Wb, nWq);
  gemm_bf16_kernel<<<dim3(S_LEN / 128, HIDN / 128), 256, 0, stream>>>(
      Xb, Wb, Qf, S_LEN, HIDN, HIDN);
  // K = X @ Wk
  f2bf_kernel<<<(nWk + 255) / 256, 256, 0, stream>>>(Wk, Wb, nWk);
  gemm_bf16_kernel<<<dim3(S_LEN / 128, (NKVH * HD) / 128), 256, 0, stream>>>(
      Xb, Wb, Kf, S_LEN, NKVH * HD, HIDN);
  // V = X @ Wv
  f2bf_kernel<<<(nWk + 255) / 256, 256, 0, stream>>>(Wv, Wb, nWk);
  gemm_bf16_kernel<<<dim3(S_LEN / 128, (NKVH * HD) / 128), 256, 0, stream>>>(
      Xb, Wb, Vf, S_LEN, NKVH * HD, HIDN);

  // RoPE
  rope_q_kernel<<<(S_LEN * NH * HD) / 256, 256, 0, stream>>>(Qf, cosb, sinb, Qbb);
  rope_k_kernel<<<(S_LEN * NKVH * HD) / 256, 256, 0, stream>>>(Kf, cosb, sinb, Krf);

  // token importance from the last LASTQ queries
  zero_kernel<<<(S_LEN + 255) / 256, 256, 0, stream>>>(TI, S_LEN);
  importance_kernel<<<dim3(LASTQ, NH), 256, 0, stream>>>(Qbb, Krf, TI);

  // block sort + per-token quant levels
  bits_kernel<<<1, 256, 0, stream>>>(TI, ratio, LEV);

  // fake-quant KV
  quant_kernel<<<NKVH * S_LEN, 128, 0, stream>>>(Krf, Vf, LEV, Kqb, Vqb);

  // causal flash attention over quantized KV
  attn_kernel<<<dim3(S_LEN / 128, NH), 256, 0, stream>>>(Qbb, Kqb, Vqb, Obb);

  // out = attn @ Wo
  f2bf_kernel<<<(nWq + 255) / 256, 256, 0, stream>>>(Wo, Wb, nWq);
  gemm_bf16_kernel<<<dim3(S_LEN / 128, HIDN / 128), 256, 0, stream>>>(
      Obb, Wb, out, S_LEN, HIDN, HIDN);
}
